// MultiAttention_45217415693129
// MI455X (gfx1250) — matrix operand order repaired and checked by reading, not compiled, not run on hardware
//
#include <hip/hip_runtime.h>
#include <hip/hip_bf16.h>
#include <cstdint>

// ---------------------------------------------------------------------------
// Problem constants
// ---------------------------------------------------------------------------
#define BB   32
#define NN   512
#define HH   8
#define DHD  48
#define EE   384          // H*DH
#define INN  768          // C*NT
#define N3E  1152         // 3*E
#define MM   (BB * NN)    // 16384
#define NTOK 24
#define OUTD 768

typedef __attribute__((ext_vector_type(16))) __bf16 v16bf;
typedef __attribute__((ext_vector_type(8)))  float  v8f;
typedef __attribute__((ext_vector_type(4)))  unsigned int u32x4;
typedef __attribute__((ext_vector_type(8)))  int i32x8;
typedef __attribute__((ext_vector_type(4)))  int i32x4;

#if __has_builtin(__builtin_amdgcn_tensor_load_to_lds)
#define HAVE_TDM 1
#else
#define HAVE_TDM 0
#endif

struct Frag32B { uint4 lo, hi; };

__device__ __forceinline__ unsigned short f2bf(float f) {
  unsigned int u = __builtin_bit_cast(unsigned int, f);
  u += 0x7FFFu + ((u >> 16) & 1u);           // round-to-nearest-even
  return (unsigned short)(u >> 16);
}

// A fragment (16x32 bf16): lane l -> row (l&15), k = kb..kb+15,
// kb = (l>>4)*16.  `p` is K-contiguous with row stride `stride` (elements).
__device__ __forceinline__ v16bf frag_a(const unsigned short* p, int row, int stride, int kb) {
  Frag32B f;
  f.lo = *(const uint4*)(p + row * stride + kb);
  f.hi = *(const uint4*)(p + row * stride + kb + 8);
  return __builtin_bit_cast(v16bf, f);
}
// B fragment (32x16 bf16): lane l -> col (l&15), k = ks..ks+15, ks=(l>>4)*16.
// `p` is stored [col][k] (transposed), K-contiguous.
__device__ __forceinline__ v16bf frag_b(const unsigned short* p, int col, int stride, int ks) {
  Frag32B f;
  f.lo = *(const uint4*)(p + col * stride + ks);
  f.hi = *(const uint4*)(p + col * stride + ks + 8);
  return __builtin_bit_cast(v16bf, f);
}
__device__ __forceinline__ v8f bfwmma(v16bf a, v16bf b, v8f c) {
  return __builtin_amdgcn_wmma_f32_16x16x32_bf16(false, a, false, b, (short)0, c, false, false);
}

#if HAVE_TDM
// Issue one TDM 2D tile load: global (row-major, `stride` elems/row, 2B elems)
// -> LDS at byte offset lds_off.  Tile = tile_w (k elems) x tile_h (rows).
// D# layout per CDNA5 ISA ch.8 (group0 128b, group1 256b; groups 2/3 zero = 2D).
__device__ __forceinline__ void tdm_load_2d(unsigned lds_off, const void* gptr,
                                            unsigned tensor_w, unsigned tensor_h,
                                            unsigned tile_w, unsigned tile_h,
                                            unsigned stride_elems) {
  unsigned long long ga = (unsigned long long)(size_t)gptr;
  unsigned long long s0 = (unsigned long long)stride_elems;
  u32x4 g0;
  g0[0] = 1u;                                               // count=1, user D#
  g0[1] = lds_off;                                          // lds_addr [63:32]
  g0[2] = (unsigned)(ga & 0xffffffffu);                     // global_addr lo
  g0[3] = (unsigned)((ga >> 32) & 0x01ffffffu) | (2u << 30);// addr hi | type=2
  i32x8 g1;
  g1[0] = (int)(1u << 16);                                  // wg_mask=0, data_size=2B
  g1[1] = (int)((tensor_w & 0xffffu) << 16);                // tensor_dim0[15:0] @63:48
  g1[2] = (int)((tensor_w >> 16) | ((tensor_h & 0xffffu) << 16));
  g1[3] = (int)((tensor_h >> 16) | (tile_w << 16));         // tile_dim0 @127:112
  g1[4] = (int)(tile_h & 0xffffu);                          // tile_dim1, tile_dim2=0
  g1[5] = (int)(s0 & 0xffffffffu);                          // dim0_stride lo32
  g1[6] = (int)((s0 >> 32) & 0xffffu);                      // dim0_stride hi16
  g1[7] = 0;
  i32x4 z4 = {0, 0, 0, 0};
  i32x8 z8 = {0, 0, 0, 0, 0, 0, 0, 0};
  __builtin_amdgcn_tensor_load_to_lds(g0, g1, z4, z4, z8, 0);
}
#endif

// ---------------------------------------------------------------------------
// Elementwise prep kernels
// ---------------------------------------------------------------------------
__global__ void cvt_f32_bf16(const float* __restrict__ src, unsigned short* __restrict__ dst,
                             int n) {
  int i = blockIdx.x * blockDim.x + threadIdx.x;
  if (i < n) dst[i] = f2bf(src[i]);
}

// W[K][N] f32 -> WT[n*K + k] bf16  (transpose + convert)
__global__ void prep_wT(const float* __restrict__ W, unsigned short* __restrict__ WT,
                        int K, int N) {
  int i = blockIdx.x * blockDim.x + threadIdx.x;
  if (i >= K * N) return;
  int n = i / K, k = i - n * K;
  WT[(size_t)n * K + k] = f2bf(W[(size_t)k * N + n]);
}

// Fused FC weight: Wcat[n][k] = k<384 ? Wfc_t[k][n] : Wfc_s[k-384][n]; bias sum.
__global__ void prep_wcat(const float* __restrict__ Wt, const float* __restrict__ Ws,
                          const float* __restrict__ bt, const float* __restrict__ bs,
                          unsigned short* __restrict__ Wcat, float* __restrict__ bcat) {
  int i = blockIdx.x * blockDim.x + threadIdx.x;
  if (i >= OUTD * INN) return;
  int n = i / INN, k = i - n * INN;
  float v = (k < EE) ? Wt[(size_t)k * OUTD + n] : Ws[(size_t)(k - EE) * OUTD + n];
  Wcat[(size_t)n * INN + k] = f2bf(v);
  if (i < OUTD) bcat[i] = bt[i] + bs[i];
}

// ---------------------------------------------------------------------------
// bf16 WMMA GEMM with TDM-staged, double-buffered LDS tiles.
//   C[M][N] = A[M][K] @ Bt[N][K]^T + bias[N]
// block 256 = 8 waves; block tile 128(M) x 128(N); wave tile 32x64 (2x4 WMMA).
// Dynamic LDS only => TDM lds_addr base is 0.  Layout (bytes):
//   [0]      bufA[2][128*32] bf16   (2 x 8 KB)
//   [16384]  bufB[2][128*32] bf16   (2 x 8 KB)
// ---------------------------------------------------------------------------
__global__ void __launch_bounds__(256)
wmma_gemm(const unsigned short* __restrict__ A, const unsigned short* __restrict__ Bt,
          const float* __restrict__ bias, float* __restrict__ C, int M, int N, int K) {
  extern __shared__ __align__(16) unsigned short smem[];   // 32 KB dynamic
  const int n0 = blockIdx.x * 128;
  const int m0 = blockIdx.y * 128;
  const int tid = threadIdx.x;
  const int wv = tid >> 5, lane = tid & 31, hl = lane >> 4, c16 = lane & 15;
  const int wm = (wv & 3) * 32, wn = (wv >> 2) * 64;

  v8f acc[2][4];
  #pragma unroll
  for (int mi = 0; mi < 2; ++mi)
    #pragma unroll
    for (int nj = 0; nj < 4; ++nj) acc[mi][nj] = v8f{0.f,0.f,0.f,0.f,0.f,0.f,0.f,0.f};

  const int ksteps = K / 32;

#if HAVE_TDM
  if (wv == 0) {  // prologue: stage k-step 0 into buffer 0 (TDM, per-wave op)
    tdm_load_2d(0u,     A  + (size_t)m0 * K, (unsigned)K, (unsigned)M, 32u, 128u, (unsigned)K);
    tdm_load_2d(16384u, Bt + (size_t)n0 * K, (unsigned)K, (unsigned)N, 32u, 128u, (unsigned)K);
  }
#endif

  for (int j = 0; j < ksteps; ++j) {
    const int cur = j & 1;
#if HAVE_TDM
    if (wv == 0) __builtin_amdgcn_s_wait_tensorcnt(0);
    __syncthreads();                       // TDM data visible; prev buffers free
    if (wv == 0 && j + 1 < ksteps) {       // prefetch next k-step into other buffer
      const int k0 = (j + 1) * 32;
      const unsigned nb = (unsigned)((j + 1) & 1) * 8192u;
      tdm_load_2d(nb,          A  + (size_t)m0 * K + k0, (unsigned)K, (unsigned)M, 32u, 128u, (unsigned)K);
      tdm_load_2d(16384u + nb, Bt + (size_t)n0 * K + k0, (unsigned)K, (unsigned)N, 32u, 128u, (unsigned)K);
    }
#else
    __syncthreads();
    {                                      // manual staging fallback
      const int k0 = j * 32;
      #pragma unroll
      for (int i = 0; i < 2; ++i) {
        int cid = i * 256 + tid;           // 512 chunks of 8 bf16
        int r = cid >> 2, kc = (cid & 3) * 8;
        *(uint4*)&smem[cur * 4096 + r * 32 + kc] =
            *(const uint4*)&A[(size_t)(m0 + r) * K + k0 + kc];
        *(uint4*)&smem[8192 + cur * 4096 + r * 32 + kc] =
            *(const uint4*)&Bt[(size_t)(n0 + r) * K + k0 + kc];
      }
    }
    __syncthreads();
#endif

    const unsigned short* lsA = smem + cur * 4096;
    const unsigned short* lsB = smem + 8192 + cur * 4096;
    v16bf a0 = frag_a(lsA, wm + c16,      32, hl * 16);
    v16bf a1 = frag_a(lsA, wm + 16 + c16, 32, hl * 16);
    #pragma unroll
    for (int nj = 0; nj < 4; ++nj) {
      v16bf bb = frag_b(lsB, wn + nj * 16 + c16, 32, hl * 16);
      acc[0][nj] = bfwmma(a0, bb, acc[0][nj]);
      acc[1][nj] = bfwmma(a1, bb, acc[1][nj]);
    }
  }

  #pragma unroll
  for (int mi = 0; mi < 2; ++mi)
    #pragma unroll
    for (int nj = 0; nj < 4; ++nj)
      #pragma unroll
      for (int r = 0; r < 8; ++r) {
        int row = m0 + wm + mi * 16 + hl * 8 + r;
        int col = n0 + wn + nj * 16 + c16;
        C[(size_t)row * N + col] = acc[mi][nj][r] + bias[col];
      }
}

// ---------------------------------------------------------------------------
// Temporal flash attention.  qkv layout: [b*512 + n][3*384] with q,k,v at
// offsets 0/384/768 and head offset h*48.  DH=48 padded to 64 with zeros.
// block 128 = 4 waves; block handles (b, h, 128 query rows); wave owns 32 rows.
// ---------------------------------------------------------------------------
__global__ void __launch_bounds__(128)
flash_temporal(const float* __restrict__ qkv, float* __restrict__ xt) {
  const int m0 = blockIdx.x * 128;
  const int h = blockIdx.y;
  const int b = blockIdx.z;
  const int tid = threadIdx.x;
  const int wv = tid >> 5, lane = tid & 31, hl = lane >> 4, c16 = lane & 15;
  const size_t rowbase = (size_t)b * NN * N3E;
  const float scale = 0.14433756729740643f;  // 1/sqrt(48)

  __shared__ __align__(16) unsigned short lsQ[128 * 64];
  __shared__ __align__(16) unsigned short lsK[32 * 64];
  __shared__ __align__(16) unsigned short lsV[48 * 32];      // stored [d][key]
  __shared__ __align__(16) unsigned short lsP[4 * 32 * 32];  // per-wave P tile

  for (int e = tid; e < 128 * 64; e += 128) {
    int r = e >> 6, d = e & 63;
    float v = (d < DHD) ? qkv[rowbase + (size_t)(m0 + r) * N3E + h * DHD + d] : 0.f;
    lsQ[r * 64 + d] = f2bf(v);
  }

  v8f o[2][3];
  float mrow[2][8], lrow[2][8];
  #pragma unroll
  for (int mi = 0; mi < 2; ++mi) {
    #pragma unroll
    for (int nj = 0; nj < 3; ++nj) o[mi][nj] = v8f{0.f,0.f,0.f,0.f,0.f,0.f,0.f,0.f};
    #pragma unroll
    for (int r = 0; r < 8; ++r) { mrow[mi][r] = -1e30f; lrow[mi][r] = 0.f; }
  }

  for (int j = 0; j < NN / 32; ++j) {
    __syncthreads();
    for (int e = tid; e < 32 * 64; e += 128) {
      int r = e >> 6, d = e & 63;
      float v = (d < DHD) ? qkv[rowbase + (size_t)(j * 32 + r) * N3E + EE + h * DHD + d] : 0.f;
      lsK[r * 64 + d] = f2bf(v);
    }
    for (int e = tid; e < 48 * 32; e += 128) {
      int d = e >> 5, key = e & 31;
      float v = qkv[rowbase + (size_t)(j * 32 + key) * N3E + 2 * EE + h * DHD + d];
      lsV[d * 32 + key] = f2bf(v);
    }
    __syncthreads();

    // S = Q @ K^T  (K-dim 64 = 2 wmma steps)
    v8f s[2][2];
    #pragma unroll
    for (int mi = 0; mi < 2; ++mi)
      #pragma unroll
      for (int nj = 0; nj < 2; ++nj) s[mi][nj] = v8f{0.f,0.f,0.f,0.f,0.f,0.f,0.f,0.f};
    #pragma unroll
    for (int ks = 0; ks < 2; ++ks) {
      v16bf a0 = frag_a(lsQ, wv * 32 + c16,      64, ks * 32 + hl * 16);
      v16bf a1 = frag_a(lsQ, wv * 32 + 16 + c16, 64, ks * 32 + hl * 16);
      v16bf b0 = frag_b(lsK, c16,      64, ks * 32 + hl * 16);
      v16bf b1 = frag_b(lsK, 16 + c16, 64, ks * 32 + hl * 16);
      s[0][0] = bfwmma(a0, b0, s[0][0]);
      s[0][1] = bfwmma(a0, b1, s[0][1]);
      s[1][0] = bfwmma(a1, b0, s[1][0]);
      s[1][1] = bfwmma(a1, b1, s[1][1]);
    }

    // online softmax (row = hl*8 + r within 16x16 tile; 16 lanes share a row)
    #pragma unroll
    for (int mi = 0; mi < 2; ++mi)
      #pragma unroll
      for (int r = 0; r < 8; ++r) {
        float s0 = s[mi][0][r] * scale;
        float s1 = s[mi][1][r] * scale;
        float mx = fmaxf(s0, s1);
        mx = fmaxf(mx, __shfl_xor(mx, 1, 32));
        mx = fmaxf(mx, __shfl_xor(mx, 2, 32));
        mx = fmaxf(mx, __shfl_xor(mx, 4, 32));
        mx = fmaxf(mx, __shfl_xor(mx, 8, 32));
        float mnew = fmaxf(mrow[mi][r], mx);
        float p0 = __expf(s0 - mnew);
        float p1 = __expf(s1 - mnew);
        float corr = __expf(mrow[mi][r] - mnew);
        float ps = p0 + p1;
        ps += __shfl_xor(ps, 1, 32);
        ps += __shfl_xor(ps, 2, 32);
        ps += __shfl_xor(ps, 4, 32);
        ps += __shfl_xor(ps, 8, 32);
        lrow[mi][r] = lrow[mi][r] * corr + ps;
        mrow[mi][r] = mnew;
        #pragma unroll
        for (int nj = 0; nj < 3; ++nj) o[mi][nj][r] *= corr;
        int prow = mi * 16 + hl * 8 + r;
        lsP[wv * 1024 + prow * 32 + c16]      = f2bf(p0);
        lsP[wv * 1024 + prow * 32 + 16 + c16] = f2bf(p1);
      }
    asm volatile("s_wait_dscnt 0" ::: "memory");

    // O += P @ V  (K-dim = 32 keys, one wmma per tile)
    {
      const unsigned short* pb = lsP + wv * 1024;
      v16bf pa0 = frag_a(pb, c16,      32, hl * 16);
      v16bf pa1 = frag_a(pb, 16 + c16, 32, hl * 16);
      #pragma unroll
      for (int nj = 0; nj < 3; ++nj) {
        v16bf vb = frag_b(lsV, nj * 16 + c16, 32, hl * 16);
        o[0][nj] = bfwmma(pa0, vb, o[0][nj]);
        o[1][nj] = bfwmma(pa1, vb, o[1][nj]);
      }
    }
  }

  #pragma unroll
  for (int mi = 0; mi < 2; ++mi)
    #pragma unroll
    for (int nj = 0; nj < 3; ++nj)
      #pragma unroll
      for (int r = 0; r < 8; ++r) {
        int row = m0 + wv * 32 + mi * 16 + hl * 8 + r;
        int col = h * DHD + nj * 16 + c16;
        xt[((size_t)b * NN + row) * EE + col] = o[mi][nj][r] / lrow[mi][r];
      }
}

// ---------------------------------------------------------------------------
// Spatial attention: per (g = b*n, h): q,k,v are (2 x 24); one wave per (g,h).
// ---------------------------------------------------------------------------
__global__ void __launch_bounds__(256)
spatial_attn(const float* __restrict__ qkv, float* __restrict__ xs) {
  const int wv = threadIdx.x >> 5, lane = threadIdx.x & 31;
  const int idx = blockIdx.x * 8 + wv;     // 0 .. B*N*H-1
  const int g = idx >> 3, h = idx & 7;
  const size_t base = (size_t)g * N3E + h * DHD;
  const int t = lane;

  float q0 = qkv[base + t];
  float q1 = qkv[base + NTOK + t];
  float s[NTOK];
  float mx = -1e30f;
  #pragma unroll
  for (int u = 0; u < NTOK; ++u) {
    float k0 = qkv[base + EE + u];
    float k1 = qkv[base + EE + NTOK + u];
    s[u] = q0 * k0 + q1 * k1;            // unscaled, per reference
    mx = fmaxf(mx, s[u]);
  }
  float sum = 0.f;
  #pragma unroll
  for (int u = 0; u < NTOK; ++u) { s[u] = __expf(s[u] - mx); sum += s[u]; }
  float a0 = 0.f, a1 = 0.f;
  #pragma unroll
  for (int u = 0; u < NTOK; ++u) {
    a0 += s[u] * qkv[base + 2 * EE + u];
    a1 += s[u] * qkv[base + 2 * EE + NTOK + u];
  }
  if (t < NTOK) {
    float inv = 1.f / sum;
    xs[(size_t)g * EE + h * DHD + t]        = a0 * inv;
    xs[(size_t)g * EE + h * DHD + NTOK + t] = a1 * inv;
  }
}

// ---------------------------------------------------------------------------
// Gating: mean over n, matvec + pair softmax, apply gate + convert to bf16.
// ---------------------------------------------------------------------------
__global__ void gate_mean(const float* __restrict__ xt, const float* __restrict__ xs,
                          float* __restrict__ mean) {
  int b = blockIdx.x;
  for (int i = 0; i < 3; ++i) {
    int col = threadIdx.x + i * 256;
    float s = 0.f;
    for (int n = 0; n < NN; ++n) {
      size_t m = (size_t)b * NN + n;
      s += (col < EE) ? xt[m * EE + col] : xs[m * EE + (col - EE)];
    }
    mean[b * 2 * EE + col] = s * (1.f / NN);
  }
}

__global__ void __launch_bounds__(384)
gate_vec(const float* __restrict__ mean, const float* __restrict__ Wts,
         const float* __restrict__ bts, float* __restrict__ gate) {
  int b = blockIdx.x, e = threadIdx.x;   // e in 0..383
  float a0 = bts[2 * e], a1 = bts[2 * e + 1];
  for (int k = 0; k < 2 * EE; ++k) {
    float mv = mean[b * 2 * EE + k];
    a0 += mv * Wts[(size_t)k * 2 * EE + 2 * e];
    a1 += mv * Wts[(size_t)k * 2 * EE + 2 * e + 1];
  }
  float mx = fmaxf(a0, a1);
  float e0 = __expf(a0 - mx), e1 = __expf(a1 - mx);
  float inv = 1.f / (e0 + e1);
  gate[b * 2 * EE + e]      = e0 * inv;   // temporal weight
  gate[b * 2 * EE + EE + e] = e1 * inv;   // spatial weight
}

__global__ void gate_apply(const float* __restrict__ xt, const float* __restrict__ xs,
                           const float* __restrict__ gate, unsigned short* __restrict__ g) {
  int i = blockIdx.x * blockDim.x + threadIdx.x;
  if (i >= MM * 2 * EE) return;
  int m = i / (2 * EE), col = i - m * (2 * EE);
  int b = m >> 9;
  float v = (col < EE) ? xt[(size_t)m * EE + col] : xs[(size_t)m * EE + (col - EE)];
  g[i] = f2bf(v * gate[b * 2 * EE + col]);
}

// ---------------------------------------------------------------------------
// Launch
// ---------------------------------------------------------------------------
extern "C" void kernel_launch(void* const* d_in, const int* in_sizes, int n_in,
                              void* d_out, int out_size, void* d_ws, size_t ws_size,
                              hipStream_t stream) {
  const float* x      = (const float*)d_in[0];
  const float* Wqkv_t = (const float*)d_in[1];
  const float* bqkv_t = (const float*)d_in[2];
  const float* Wqkv_s = (const float*)d_in[3];
  const float* bqkv_s = (const float*)d_in[4];
  const float* Wts    = (const float*)d_in[5];
  const float* bts    = (const float*)d_in[6];
  const float* Wfc_t  = (const float*)d_in[7];
  const float* bfc_t  = (const float*)d_in[8];
  const float* Wfc_s  = (const float*)d_in[9];
  const float* bfc_s  = (const float*)d_in[10];
  float* out = (float*)d_out;

  char* ws = (char*)d_ws;
  unsigned short* xf   = (unsigned short*)(ws);                 // 16384x768 bf16
  unsigned short* wtt  = (unsigned short*)(ws + 25165824);      // 1152x768 bf16
  unsigned short* wtss = (unsigned short*)(ws + 26935296);      // 1152x768 bf16
  unsigned short* wcat = (unsigned short*)(ws + 28704768);      // 768x768 bf16
  float* bcat          = (float*)(ws + 29884416);               // 768 f32
  float* qkvt          = (float*)(ws + 29887488);               // 16384x1152 f32
  float* qkvs          = (float*)(ws + 105384960);              // 16384x1152 f32
  float* xt            = (float*)(ws + 180882432);              // 16384x384 f32
  float* xs            = (float*)(ws + 206048256);              // 16384x384 f32
  float* meanb         = (float*)(ws + 231214080);              // 32x768 f32
  float* gate          = (float*)(ws + 231312384);              // 32x768 f32
  unsigned short* gbuf = (unsigned short*)(ws + 231410688);     // 16384x768 bf16

  // prep
  cvt_f32_bf16<<<(MM * INN) / 256, 256, 0, stream>>>(x, xf, MM * INN);
  prep_wT<<<(INN * N3E) / 256, 256, 0, stream>>>(Wqkv_t, wtt, INN, N3E);
  prep_wT<<<(INN * N3E) / 256, 256, 0, stream>>>(Wqkv_s, wtss, INN, N3E);
  prep_wcat<<<(OUTD * INN) / 256, 256, 0, stream>>>(Wfc_t, Wfc_s, bfc_t, bfc_s, wcat, bcat);

  // QKV projections (WMMA + TDM double-buffered staging)
  const size_t lds_bytes = 32768;
  dim3 gq(N3E / 128, MM / 128);
  wmma_gemm<<<gq, 256, lds_bytes, stream>>>(xf, wtt,  bqkv_t, qkvt, MM, N3E, INN);
  wmma_gemm<<<gq, 256, lds_bytes, stream>>>(xf, wtss, bqkv_s, qkvs, MM, N3E, INN);

  // attentions
  flash_temporal<<<dim3(NN / 128, HH, BB), 128, 0, stream>>>(qkvt, xt);
  spatial_attn<<<(MM * HH) / 8, 256, 0, stream>>>(qkvs, xs);

  // gating
  gate_mean<<<BB, 256, 0, stream>>>(xt, xs, meanb);
  gate_vec<<<BB, 384, 0, stream>>>(meanb, Wts, bts, gate);
  gate_apply<<<(MM * 2 * EE) / 256, 256, 0, stream>>>(xt, xs, gate, gbuf);

  // fused output GEMM: [g_t | g_s] @ [Wfc_t ; Wfc_s] + (bfc_t + bfc_s)
  dim3 gf(OUTD / 128, MM / 128);
  wmma_gemm<<<gf, 256, lds_bytes, stream>>>(gbuf, wcat, bcat, out, MM, OUTD, INN);
}
